// PartialEncoderEDDI_6846177870200
// MI455X (gfx1250) — matrix operand order repaired and checked by reading, not compiled, not run on hardware
//
#include <hip/hip_runtime.h>
#include <hip/hip_bf16.h>

typedef __attribute__((ext_vector_type(16))) __bf16 v16bf;
typedef __attribute__((ext_vector_type(8)))  __bf16 v8bf;
typedef __attribute__((ext_vector_type(8)))  float  v8f;

#define EPSF 1e-5f
#define B_   64
#define D_   20000
#define C_   16
#define H_   64
#define EH_  128
#define L2_  64      // 2*latent
#define WAVES 4
#define NTILES (D_/16)   // 1250, exact

#define LDS_FENCE() asm volatile("s_wait_dscnt 0" ::: "memory")

// ---------------- zero workspace accumulators ----------------
__global__ void zero_ws_kernel(float* __restrict__ p, int n) {
  int t = blockIdx.x * blockDim.x + threadIdx.x;
  if (t < n) p[t] = 0.f;
}

// ---------------- fe_proj[d,k] = b1[k] + fe[d,:] @ W1[1:,k] ----------------
__global__ void feproj_kernel(const float* __restrict__ fe,
                              const float* __restrict__ W1,
                              const float* __restrict__ b1,
                              float* __restrict__ fe_proj) {
  int t = blockIdx.x * blockDim.x + threadIdx.x;
  if (t >= D_ * H_) return;
  int d = t >> 6, k = t & 63;
  float acc = b1[k];
  const float* fr = fe + (size_t)d * C_;
#pragma unroll
  for (int c = 0; c < C_; ++c) acc = fmaf(fr[c], W1[(1 + c) * H_ + k], acc);
  fe_proj[t] = acc;
}

// ---------------- main: axpy + LN1 + ReLU -> WMMA(bf16) -> LN2 + ReLU -> masked pool ----------------
__global__ void __launch_bounds__(WAVES * 32)
pointnet_pool_kernel(const float* __restrict__ x,
                     const int*   __restrict__ mask,
                     const float* __restrict__ fe_proj,
                     const float* __restrict__ W1,   // row 0 used
                     const float* __restrict__ g1,  const float* __restrict__ be1,
                     const float* __restrict__ W2,
                     const float* __restrict__ b2,
                     const float* __restrict__ g2,  const float* __restrict__ be2,
                     float* __restrict__ g_partial, float* __restrict__ g_count) {
  __shared__ float  s_acc[C_];
  __shared__ float  s_cnt;
  // h1 stored directly in WMMA A-fragment order: [k-chunk][lane][elem]
  __shared__ __bf16 s_h1frag[WAVES][2][32][16] __attribute__((aligned(32)));
  __shared__ float  s_out[WAVES][16][C_] __attribute__((aligned(16)));

  const int tid  = threadIdx.x;
  const int wave = tid >> 5;
  const int lane = tid & 31;
  const int b    = blockIdx.y;

  if (tid < C_) s_acc[tid] = 0.f;
  if (tid == C_) s_cnt = 0.f;
  __syncthreads();

  const int mrow  = lane & 15;   // M (A) or N (B/D) index
  const int khalf = lane >> 4;   // K/M half owned by this lane (WMMA layout)
  const int r2    = lane >> 1;   // row 0..15 (phases 1/2/5: 2 lanes per row)
  const int kh    = lane & 1;    // which 32-wide K half (ph1/2) / 8-wide C half (ph5)

  // ---- hoist loop-invariant B fragments (W2, 32x16 bf16 per chunk) into VGPRs
  v16bf bfrag[2];
#pragma unroll
  for (int cch = 0; cch < 2; ++cch) {
#pragma unroll
    for (int j = 0; j < 8; ++j) {
      int kk = cch * 32 + (khalf << 4) + (j << 1);
      bfrag[cch][2 * j]     = (__bf16)W2[kk * C_ + mrow];
      bfrag[cch][2 * j + 1] = (__bf16)W2[(kk + 1) * C_ + mrow];
    }
  }

  // ---- hoist per-lane constant slices (loop-invariant; keeps them out of the
  //      fenced loop so the asm "memory" clobber can't force LDS/global reloads)
  float wv[32], gv1[32], bv1[32];
  {
    const float4* w4  = (const float4*)(W1 + kh * 32);
    const float4* g4  = (const float4*)(g1 + kh * 32);
    const float4* be4 = (const float4*)(be1 + kh * 32);
#pragma unroll
    for (int j = 0; j < 8; ++j) {
      float4 a = w4[j], g = g4[j], e = be4[j];
      wv[4 * j + 0] = a.x; wv[4 * j + 1] = a.y; wv[4 * j + 2] = a.z; wv[4 * j + 3] = a.w;
      gv1[4 * j + 0] = g.x; gv1[4 * j + 1] = g.y; gv1[4 * j + 2] = g.z; gv1[4 * j + 3] = g.w;
      bv1[4 * j + 0] = e.x; bv1[4 * j + 1] = e.y; bv1[4 * j + 2] = e.z; bv1[4 * j + 3] = e.w;
    }
  }
  float b2v[8], g2v[8], be2v[8];
  {
    const float4* p0 = (const float4*)(b2 + kh * 8);
    const float4* p1 = (const float4*)(g2 + kh * 8);
    const float4* p2 = (const float4*)(be2 + kh * 8);
    float4 a0 = p0[0], a1 = p0[1], c0 = p1[0], c1 = p1[1], d0v = p2[0], d1v = p2[1];
    b2v[0] = a0.x; b2v[1] = a0.y; b2v[2] = a0.z; b2v[3] = a0.w;
    b2v[4] = a1.x; b2v[5] = a1.y; b2v[6] = a1.z; b2v[7] = a1.w;
    g2v[0] = c0.x; g2v[1] = c0.y; g2v[2] = c0.z; g2v[3] = c0.w;
    g2v[4] = c1.x; g2v[5] = c1.y; g2v[6] = c1.z; g2v[7] = c1.w;
    be2v[0] = d0v.x; be2v[1] = d0v.y; be2v[2] = d0v.z; be2v[3] = d0v.w;
    be2v[4] = d1v.x; be2v[5] = d1v.y; be2v[6] = d1v.z; be2v[7] = d1v.w;
  }

  float accv[8];
#pragma unroll
  for (int e = 0; e < 8; ++e) accv[e] = 0.f;
  float cnt = 0.f;

  const int tileStride = gridDim.x * WAVES;
  for (int tile = blockIdx.x * WAVES + wave; tile < NTILES; tile += tileStride) {
    const int d0 = tile * 16;
    const int tnext = tile + tileStride;
    if (tnext < NTILES) __builtin_prefetch(fe_proj + (size_t)tnext * 16 * H_, 0, 1);

    // issue scalar global loads early (pinned before the first fence)
    const float xv   = x[(size_t)b * D_ + d0 + r2];
    const float mval = (float)mask[(size_t)b * D_ + d0 + r2];

    // ---- phase 1: pre1 = x * W1row0 + fe_proj, kept in registers (32 per lane)
    float pv[32];
    {
      const float4* fp = (const float4*)(fe_proj + (size_t)(d0 + r2) * H_ + kh * 32);
#pragma unroll
      for (int j = 0; j < 8; ++j) {
        float4 f = fp[j];
        pv[4 * j + 0] = fmaf(xv, wv[4 * j + 0], f.x);
        pv[4 * j + 1] = fmaf(xv, wv[4 * j + 1], f.y);
        pv[4 * j + 2] = fmaf(xv, wv[4 * j + 2], f.z);
        pv[4 * j + 3] = fmaf(xv, wv[4 * j + 3], f.w);
      }
    }

    // ---- phase 2: LayerNorm(H=64) across lane pair, ReLU, bf16 fragment store
    {
      float s = 0.f, ss = 0.f;
#pragma unroll
      for (int e = 0; e < 32; ++e) { s += pv[e]; ss += pv[e] * pv[e]; }
      s  += __shfl_xor(s, 1);
      ss += __shfl_xor(ss, 1);
      const float mean = s * (1.f / H_);
      const float var  = ss * (1.f / H_) - mean * mean;
      const float rs   = rsqrtf(var + EPSF);
#pragma unroll
      for (int e = 0; e < 32; ++e)
        pv[e] = fmaxf((pv[e] - mean) * rs * gv1[e] + bv1[e], 0.f);
      // scatter lane's 32 normalized K-values into A-fragment order:
      // 4 groups of 8 consecutive K -> one 16B LDS store each
#pragma unroll
      for (int g = 0; g < 4; ++g) {
        const int L  = r2 + ((g & 1) << 4);              // dest fragment lane
        const int i0 = (g >> 1) << 3;                    // dest elem base (0 or 8)
        const int rb = ((g & 1) << 4) + ((g >> 1) << 3); // source K base in pv
        v8bf t;
#pragma unroll
        for (int e = 0; e < 8; ++e) t[e] = (__bf16)pv[rb + e];
        *(v8bf*)&s_h1frag[wave][kh][L][i0] = t;
      }
    }
    LDS_FENCE();

    // ---- phase 3: D = h1[16x64] @ W2[64x16], two chained bf16 WMMAs
    v8f cfrag = {};
#pragma unroll
    for (int cch = 0; cch < 2; ++cch) {
      v16bf afrag = *(const v16bf*)&s_h1frag[wave][cch][lane][0];
      cfrag = __builtin_amdgcn_wmma_f32_16x16x32_bf16(
          false, afrag, false, bfrag[cch], (short)0, cfrag, false, false);
    }

    // ---- phase 4: spill D tile to LDS (VGPR r: M=r / r+8)
#pragma unroll
    for (int rr = 0; rr < 8; ++rr) s_out[wave][rr + (khalf << 3)][mrow] = cfrag[rr];
    LDS_FENCE();

    // ---- phase 5: +b2, LayerNorm(C=16) across lane pair, ReLU, masked pool
    {
      const float4* o4 = (const float4*)&s_out[wave][r2][kh * 8];
      float4 f0 = o4[0], f1 = o4[1];
      float ov[8];
      ov[0] = f0.x + b2v[0]; ov[1] = f0.y + b2v[1]; ov[2] = f0.z + b2v[2]; ov[3] = f0.w + b2v[3];
      ov[4] = f1.x + b2v[4]; ov[5] = f1.y + b2v[5]; ov[6] = f1.z + b2v[6]; ov[7] = f1.w + b2v[7];
      float s = 0.f, ss = 0.f;
#pragma unroll
      for (int e = 0; e < 8; ++e) { s += ov[e]; ss += ov[e] * ov[e]; }
      s  += __shfl_xor(s, 1);
      ss += __shfl_xor(ss, 1);
      const float mean = s * (1.f / C_);
      const float var  = ss * (1.f / C_) - mean * mean;
      const float rs   = rsqrtf(var + EPSF);
      if (kh == 0) cnt += mval;
#pragma unroll
      for (int e = 0; e < 8; ++e) {
        float val = fmaxf((ov[e] - mean) * rs * g2v[e] + be2v[e], 0.f);
        accv[e] = fmaf(mval, val, accv[e]);
      }
    }
  }

  // ---- block-level pooling reduction
#pragma unroll
  for (int e = 0; e < 8; ++e) atomicAdd(&s_acc[(kh << 3) + e], accv[e]);
  if (kh == 0) atomicAdd(&s_cnt, cnt);
  __syncthreads();
  if (tid < C_) atomicAdd(&g_partial[b * C_ + tid], s_acc[tid]);
  if (tid == C_) atomicAdd(&g_count[b], s_cnt);
}

// ---------------- finalize: mean pool + encoder MLP (16->128->64) ----------------
__global__ void __launch_bounds__(EH_)
encoder_kernel(const float* __restrict__ g_partial, const float* __restrict__ g_count,
               const float* __restrict__ We1, const float* __restrict__ bb1,
               const float* __restrict__ gg1, const float* __restrict__ bbe1,
               const float* __restrict__ We2, const float* __restrict__ bb2,
               const float* __restrict__ gg2, const float* __restrict__ bbe2,
               float* __restrict__ out) {
  __shared__ float s_c[C_], s_e1[EH_], s_red[4];
  const int b = blockIdx.x, t = threadIdx.x;
  if (t < C_) {
    float n = fmaxf(g_count[b], 1.f);
    s_c[t] = g_partial[b * C_ + t] / n;
  }
  if (t < 4) s_red[t] = 0.f;
  __syncthreads();

  // layer 1: 16 -> 128, LN over 128, ReLU
  float v = bb1[t];
#pragma unroll
  for (int c = 0; c < C_; ++c) v = fmaf(s_c[c], We1[c * EH_ + t], v);
  atomicAdd(&s_red[0], v);
  atomicAdd(&s_red[1], v * v);
  __syncthreads();
  {
    float mean = s_red[0] * (1.f / EH_);
    float var  = s_red[1] * (1.f / EH_) - mean * mean;
    float rs   = rsqrtf(var + EPSF);
    s_e1[t] = fmaxf((v - mean) * rs * gg1[t] + bbe1[t], 0.f);
  }
  __syncthreads();

  // layer 2: 128 -> 64, LN over 64, ReLU, split mu/logvar
  float v2 = 0.f;
  if (t < L2_) {
    v2 = bb2[t];
    for (int k = 0; k < EH_; ++k) v2 = fmaf(s_e1[k], We2[k * L2_ + t], v2);
    atomicAdd(&s_red[2], v2);
    atomicAdd(&s_red[3], v2 * v2);
  }
  __syncthreads();
  if (t < L2_) {
    float m2  = s_red[2] * (1.f / L2_);
    float var = s_red[3] * (1.f / L2_) - m2 * m2;
    float rs  = rsqrtf(var + EPSF);
    float val = fmaxf((v2 - m2) * rs * gg2[t] + bbe2[t], 0.f);
    if (t < 32) out[b * 32 + t] = val;                  // mu
    else        out[B_ * 32 + b * 32 + (t - 32)] = val; // logvar
  }
}

extern "C" void kernel_launch(void* const* d_in, const int* in_sizes, int n_in,
                              void* d_out, int out_size, void* d_ws, size_t ws_size,
                              hipStream_t stream) {
  const float* x    = (const float*)d_in[0];
  const int*   mask = (const int*)  d_in[1];
  const float* fe   = (const float*)d_in[2];
  const float* W1   = (const float*)d_in[3];
  const float* b1   = (const float*)d_in[4];
  const float* g1   = (const float*)d_in[5];
  const float* be1  = (const float*)d_in[6];
  const float* W2   = (const float*)d_in[7];
  const float* b2   = (const float*)d_in[8];
  const float* g2   = (const float*)d_in[9];
  const float* be2  = (const float*)d_in[10];
  const float* We1  = (const float*)d_in[11];
  const float* bb1  = (const float*)d_in[12];
  const float* gg1  = (const float*)d_in[13];
  const float* bbe1 = (const float*)d_in[14];
  const float* We2  = (const float*)d_in[15];
  const float* bb2  = (const float*)d_in[16];
  const float* gg2  = (const float*)d_in[17];
  const float* bbe2 = (const float*)d_in[18];
  float* out = (float*)d_out;

  float* fe_proj = (float*)d_ws;                 // D*H floats = 5.12 MB
  float* partial = fe_proj + (size_t)D_ * H_;    // B*C floats
  float* count   = partial + B_ * C_;            // B floats

  zero_ws_kernel<<<(B_ * C_ + B_ + 255) / 256, 256, 0, stream>>>(partial, B_ * C_ + B_);
  feproj_kernel<<<(D_ * H_ + 255) / 256, 256, 0, stream>>>(fe, W1, b1, fe_proj);
  dim3 grid(80, B_);
  pointnet_pool_kernel<<<grid, WAVES * 32, 0, stream>>>(
      x, mask, fe_proj, W1, g1, be1, W2, b2, g2, be2, partial, count);
  encoder_kernel<<<B_, EH_, 0, stream>>>(
      partial, count, We1, bb1, gg1, bbe1, We2, bb2, gg2, bbe2, out);
}
